// QiskitHybridDQN_70377334112274
// MI455X (gfx1250) — matrix operand order repaired and checked by reading, not compiled, not run on hardware
//
#include <hip/hip_runtime.h>
#include <hip/hip_bf16.h>
#include <hip/hip_fp16.h>

typedef __attribute__((ext_vector_type(16))) _Float16 v16h;
typedef __attribute__((ext_vector_type(8)))  float    v8f;
typedef __attribute__((ext_vector_type(4)))  float    v4f;

#define NB 1024          // batch
#define N_QUBITS 6
#define N_ACTIONS 6

union AFrag { v16h v; _Float16 h[16]; v4f f4[2]; };

// ---------------------------------------------------------------------------
// Pack f32 weights [N][K] into WMMA B-fragment order (f16).
// Layout per (kstep s, ntile t): 32 lanes x 8 dwords.
//   lane l (0..31): column n = t*16 + (l&15)
//   dword v (0..7): K pair k0 = s*32 + (l<16?0:8) + 2v + (v<4?0:8)
// ntiles is always a power of two here -> pass log2(ntiles).
// ---------------------------------------------------------------------------
__global__ void pack_w_kernel(const float* __restrict__ w, _Float16* __restrict__ wp,
                              int K, int ntShift) {
    int total = (K >> 5) << (ntShift + 8);           // dwords
    int i = blockIdx.x * blockDim.x + threadIdx.x;
    if (i >= total) return;
    int v  = i & 7;
    int l  = (i >> 3) & 31;
    int st = i >> 8;                                 // s*ntiles + t
    int s = st >> ntShift;
    int t = st & ((1 << ntShift) - 1);
    int n  = (t << 4) + (l & 15);
    int k0 = (s << 5) + ((l < 16) ? 0 : 8) + 2 * v + ((v < 4) ? 0 : 8);
    wp[(size_t)i * 2 + 0] = (_Float16)w[(size_t)n * K + k0];
    wp[(size_t)i * 2 + 1] = (_Float16)w[(size_t)n * K + k0 + 1];
}

// ---------------------------------------------------------------------------
// Implicit-GEMM convolution with WMMA f16 (f32 accum), fused bias + ReLU.
// One wave computes a 16-pixel x (NTILES*16)-channel tile; NTILES is static.
// K = IC*KH*KW (multiple of 32), M = B*OH*OW (multiple of 16*8).
// ---------------------------------------------------------------------------
template <bool USEF32, int NTILES>
__global__ void conv_wmma_kernel(const float* __restrict__ inF32,
                                 const _Float16* __restrict__ inF16,
                                 const _Float16* __restrict__ wp,
                                 const float* __restrict__ bias,
                                 _Float16* __restrict__ out,
                                 int Bn, int IC, int IH, int IW,
                                 int OC, int OH, int OW,
                                 int KH, int KW, int stride) {
    __shared__ int sDelta[1024];                     // K <= 576 here
    const int K = IC * KH * KW;
    for (int k = threadIdx.x; k < K; k += blockDim.x) {
        int ic = k / (KH * KW);
        int r  = k % (KH * KW);
        int ky = r / KW, kx = r % KW;
        sDelta[k] = ic * IH * IW + ky * IW + kx;
    }
    __syncthreads();

    const int wave = blockIdx.x * (blockDim.x >> 5) + (threadIdx.x >> 5);
    const int lane = threadIdx.x & 31;
    const int OHOW = OH * OW;
    const int M = Bn * OHOW;
    if (wave * 16 >= M) return;

    const int laneLo = lane & 15;
    const int mRow = wave * 16 + laneLo;
    const int bIdx = mRow / OHOW;
    const int rm   = mRow - bIdx * OHOW;
    const int oy = rm / OW, ox = rm - (rm / OW) * OW;
    const size_t inBase = ((size_t)bIdx * IC) * IH * IW + (size_t)(oy * stride) * IW + ox * stride;

    const int ksteps = K >> 5;
    const int kHiOff = (lane < 16) ? 0 : 8;
    const size_t bLaneOff = (size_t)lane * 16;       // halves

    v8f acc[NTILES];
#pragma unroll
    for (int t = 0; t < NTILES; ++t) { v8f z = {}; acc[t] = z; }

    for (int s = 0; s < ksteps; ++s) {
        const _Float16* wpS = wp + ((size_t)s * NTILES * 32) * 16;
        if (s + 1 < ksteps)
            __builtin_prefetch(wpS + (size_t)NTILES * 512 + bLaneOff, 0, 1);
        AFrag a;
#pragma unroll
        for (int v = 0; v < 8; ++v) {
            int klo = s * 32 + 2 * v + ((v < 4) ? 0 : 8) + kHiOff;
            int d0 = sDelta[klo], d1 = sDelta[klo + 1];
            float x0, x1;
            if (USEF32) { x0 = inF32[inBase + d0]; x1 = inF32[inBase + d1]; }
            else        { x0 = (float)inF16[inBase + d0]; x1 = (float)inF16[inBase + d1]; }
            a.h[2 * v]     = (_Float16)x0;
            a.h[2 * v + 1] = (_Float16)x1;
        }
#pragma unroll
        for (int t = 0; t < NTILES; ++t) {
            const v16h bf = *(const v16h*)(wpS + (size_t)t * 512 + bLaneOff);
            acc[t] = __builtin_amdgcn_wmma_f32_16x16x32_f16(
                false, a.v, false, bf, (short)0, acc[t], false, false);
        }
    }

    float bb[NTILES];
#pragma unroll
    for (int t = 0; t < NTILES; ++t) bb[t] = bias[t * 16 + laneLo];

    // incremental decomposition of 8 consecutive output rows (1 div total)
    int m  = wave * 16 + kHiOff;
    int b2 = m / OHOW;
    int rr = m - b2 * OHOW;
#pragma unroll
    for (int r = 0; r < 8; ++r) {
        size_t outOff = ((size_t)b2 * OC) * OHOW + rr;
#pragma unroll
        for (int t = 0; t < NTILES; ++t) {
            int oc = t * 16 + laneLo;
            float val = fmaxf(acc[t][r] + bb[t], 0.f);
            out[outOff + (size_t)oc * OHOW] = (_Float16)val;
        }
        if (++rr == OHOW) { rr = 0; ++b2; }
    }
}

// ---------------------------------------------------------------------------
// Dense GEMM: out[M][N] = relu(A[M][K] (f16) * W^T + bias), WMMA f16.
// One wave: 16 rows x (NT*16) cols. ngroups = N/(16*NT) is a power of two.
// ---------------------------------------------------------------------------
template <int NT>
__global__ void gemm_wmma_kernel(const _Float16* __restrict__ A,
                                 const _Float16* __restrict__ wp,
                                 const float* __restrict__ bias,
                                 _Float16* __restrict__ out,
                                 int M, int N, int K, int ngShift) {
    const int wave = blockIdx.x * (blockDim.x >> 5) + (threadIdx.x >> 5);
    const int lane = threadIdx.x & 31;
    const int mtile = wave >> ngShift;
    const int ng    = wave & ((1 << ngShift) - 1);
    if (mtile * 16 >= M) return;

    const int laneLo = lane & 15;
    const int mRow = mtile * 16 + laneLo;
    const _Float16* arow = A + (size_t)mRow * K;
    const int ntilesTotal = N >> 4;
    const int ksteps = K >> 5;
    const int kHiOff = (lane < 16) ? 0 : 8;
    const size_t bLaneOff = (size_t)lane * 16;       // halves
    const _Float16* wpG = wp + ((size_t)ng * NT * 32) * 16;
    const size_t wpStride = (size_t)ntilesTotal * 512;   // halves per kstep

    v8f acc[NT];
#pragma unroll
    for (int t = 0; t < NT; ++t) { v8f z = {}; acc[t] = z; }

    for (int s = 0; s < ksteps; ++s) {
        const _Float16* wpS = wpG + (size_t)s * wpStride;
        if (s + 1 < ksteps)
            __builtin_prefetch(wpS + wpStride + bLaneOff, 0, 1);
        AFrag a;
        int kb = s * 32 + kHiOff;
        a.f4[0] = *(const v4f*)(arow + kb);          // k = kb .. kb+7
        a.f4[1] = *(const v4f*)(arow + kb + 16);     // k = kb+16 .. kb+23
#pragma unroll
        for (int t = 0; t < NT; ++t) {
            const v16h bf = *(const v16h*)(wpS + (size_t)t * 512 + bLaneOff);
            acc[t] = __builtin_amdgcn_wmma_f32_16x16x32_f16(
                false, a.v, false, bf, (short)0, acc[t], false, false);
        }
    }

    const int rowBase = mtile * 16 + kHiOff;
#pragma unroll
    for (int t = 0; t < NT; ++t) {
        int col = (ng * NT + t) * 16 + laneLo;
        float bb = bias[col];
#pragma unroll
        for (int r = 0; r < 8; ++r) {
            float val = fmaxf(acc[t][r] + bb, 0.f);
            out[(size_t)(rowBase + r) * N + col] = (_Float16)val;
        }
    }
}

// ---------------------------------------------------------------------------
// Heads: value = relu_hidden @ val2_w + b ;  qin = tanh(feat @ adv_w^T + b)
// One thread per batch element.
// ---------------------------------------------------------------------------
__global__ void heads_kernel(const _Float16* __restrict__ feat,   // [B][512]
                             const _Float16* __restrict__ hid,    // [B][128] (post-relu)
                             const float* __restrict__ val2_w,    // [128]
                             const float* __restrict__ val2_b,    // [1]
                             const float* __restrict__ adv_w,     // [6][512]
                             const float* __restrict__ adv_b,     // [6]
                             float* __restrict__ value,           // [B]
                             float* __restrict__ qin) {           // [B][6]
    int b = blockIdx.x * blockDim.x + threadIdx.x;
    if (b >= NB) return;
    const _Float16* f = feat + (size_t)b * 512;
    const _Float16* h = hid  + (size_t)b * 128;
    float v = val2_b[0];
    for (int j = 0; j < 128; ++j) v += (float)h[j] * val2_w[j];
    value[b] = v;
    for (int i = 0; i < N_QUBITS; ++i) {
        float s = adv_b[i];
        const float* w = adv_w + (size_t)i * 512;
        for (int k = 0; k < 512; ++k) s += (float)f[k] * w[k];
        qin[b * N_QUBITS + i] = tanhf(s);
    }
}

// ---------------------------------------------------------------------------
// 6-qubit statevector simulation: one wave per batch element,
// 2 complex amplitudes per lane (idx = lane*2 + j; qubit q -> bit (5-q)).
// ---------------------------------------------------------------------------
__device__ inline void gate_cx(float re[2], float im[2], int lane, int c, int t) {
    int pc = 5 - c, pt = 5 - t;                      // pc >= 1 in this circuit
    int ctrl = (lane >> (pc - 1)) & 1;
    if (pt == 0) {
        if (ctrl) {
            float tr = re[0]; re[0] = re[1]; re[1] = tr;
            float ti = im[0]; im[0] = im[1]; im[1] = ti;
        }
    } else {
        int m = 1 << (pt - 1);
#pragma unroll
        for (int j = 0; j < 2; ++j) {
            float orr = __shfl_xor(re[j], m, 32);
            float oii = __shfl_xor(im[j], m, 32);
            if (ctrl) { re[j] = orr; im[j] = oii; }
        }
    }
}

__device__ inline void gate_ry(float re[2], float im[2], int lane, int q, float c, float s) {
    int p = 5 - q;
    if (p == 0) {
        float nr0 = c * re[0] - s * re[1], ni0 = c * im[0] - s * im[1];
        float nr1 = s * re[0] + c * re[1], ni1 = s * im[0] + c * im[1];
        re[0] = nr0; im[0] = ni0; re[1] = nr1; im[1] = ni1;
    } else {
        int m = 1 << (p - 1);
        int bit = (lane >> (p - 1)) & 1;
#pragma unroll
        for (int j = 0; j < 2; ++j) {
            float orr = __shfl_xor(re[j], m, 32);
            float oii = __shfl_xor(im[j], m, 32);
            if (!bit) { re[j] = c * re[j] - s * orr; im[j] = c * im[j] - s * oii; }
            else      { re[j] = s * orr + c * re[j]; im[j] = s * oii + c * im[j]; }
        }
    }
}

__device__ inline void gate_rz(float re[2], float im[2], int lane, int q, float cp, float sp) {
    int p = 5 - q;
#pragma unroll
    for (int j = 0; j < 2; ++j) {
        int idx = lane * 2 + j;
        int bit = (idx >> p) & 1;
        float sg = bit ? sp : -sp;                   // e^{-i phi} on |0>, e^{+i phi} on |1>
        float nr = re[j] * cp - im[j] * sg;
        float ni = im[j] * cp + re[j] * sg;
        re[j] = nr; im[j] = ni;
    }
}

__global__ void quantum_kernel(const float* __restrict__ qin,     // [B][6]
                               const float* __restrict__ qp,      // [24]
                               const float* __restrict__ qout_w,  // [6][6]
                               const float* __restrict__ qout_b,  // [6]
                               const float* __restrict__ value,   // [B]
                               float* __restrict__ out) {         // [B][6]
    int wave = blockIdx.x * (blockDim.x >> 5) + (threadIdx.x >> 5);
    int lane = threadIdx.x & 31;
    if (wave >= NB) return;

    float c6[N_QUBITS], s6[N_QUBITS];
#pragma unroll
    for (int q = 0; q < N_QUBITS; ++q) {
        float half = 0.5f * qin[wave * N_QUBITS + q];
        c6[q] = cosf(half); s6[q] = sinf(half);
    }
    float re[2], im[2];
#pragma unroll
    for (int j = 0; j < 2; ++j) {
        int idx = lane * 2 + j;
        float p = 1.f;
#pragma unroll
        for (int q = 0; q < N_QUBITS; ++q)
            p *= ((idx >> (5 - q)) & 1) ? s6[q] : c6[q];
        re[j] = p; im[j] = 0.f;
    }

    int w = 0;
#pragma unroll
    for (int layer = 0; layer < 2; ++layer) {
        gate_cx(re, im, lane, 0, 1);
        gate_cx(re, im, lane, 2, 3);
        gate_cx(re, im, lane, 4, 5);
        gate_cx(re, im, lane, 1, 2);
        gate_cx(re, im, lane, 3, 4);
#pragma unroll
        for (int q = 0; q < N_QUBITS; ++q) {
            float th = 0.5f * qp[w++];
            gate_ry(re, im, lane, q, cosf(th), sinf(th));
            float ph = 0.5f * qp[w++];
            gate_rz(re, im, lane, q, cosf(ph), sinf(ph));
        }
    }

    float p0 = re[0] * re[0] + im[0] * im[0];
    float p1 = re[1] * re[1] + im[1] * im[1];
    float z[N_QUBITS];
#pragma unroll
    for (int q = 0; q < N_QUBITS; ++q) {
        int p = 5 - q;
        if (p == 0) z[q] = p0 - p1;
        else {
            int bit = (lane >> (p - 1)) & 1;
            z[q] = bit ? -(p0 + p1) : (p0 + p1);
        }
    }
#pragma unroll
    for (int off = 16; off; off >>= 1)
#pragma unroll
        for (int q = 0; q < N_QUBITS; ++q)
            z[q] += __shfl_xor(z[q], off, 32);

    if (lane == 0) {
        float adv[N_ACTIONS], mean = 0.f;
#pragma unroll
        for (int i = 0; i < N_ACTIONS; ++i) {
            float s = qout_b[i];
#pragma unroll
            for (int q = 0; q < N_QUBITS; ++q) s += z[q] * qout_w[i * N_QUBITS + q];
            adv[i] = s; mean += s;
        }
        mean *= (1.f / N_ACTIONS);
        float v = value[wave];
#pragma unroll
        for (int i = 0; i < N_ACTIONS; ++i)
            out[wave * N_ACTIONS + i] = v + adv[i] - mean;
    }
}

// ---------------------------------------------------------------------------
extern "C" void kernel_launch(void* const* d_in, const int* in_sizes, int n_in,
                              void* d_out, int out_size, void* d_ws, size_t ws_size,
                              hipStream_t stream) {
    const float* x       = (const float*)d_in[0];
    const float* conv1_w = (const float*)d_in[1];
    const float* conv1_b = (const float*)d_in[2];
    const float* conv2_w = (const float*)d_in[3];
    const float* conv2_b = (const float*)d_in[4];
    const float* conv3_w = (const float*)d_in[5];
    const float* conv3_b = (const float*)d_in[6];
    const float* fc_w    = (const float*)d_in[7];
    const float* fc_b    = (const float*)d_in[8];
    const float* val1_w  = (const float*)d_in[9];
    const float* val1_b  = (const float*)d_in[10];
    const float* val2_w  = (const float*)d_in[11];
    const float* val2_b  = (const float*)d_in[12];
    const float* adv_w   = (const float*)d_in[13];
    const float* adv_b   = (const float*)d_in[14];
    const float* qparams = (const float*)d_in[15];
    const float* qout_w  = (const float*)d_in[16];
    const float* qout_b  = (const float*)d_in[17];
    float* out = (float*)d_out;
    char* ws = (char*)d_ws;

    size_t o = 0;
    auto alloc = [&](size_t bytes) { size_t r = o; o += (bytes + 255) & ~(size_t)255; return r; };
    size_t wp1  = alloc((size_t)256 * 32 * 2);
    size_t wp2  = alloc((size_t)512 * 64 * 2);
    size_t wp3  = alloc((size_t)576 * 64 * 2);
    size_t wpf  = alloc((size_t)3136 * 512 * 2);
    size_t wpv  = alloc((size_t)512 * 128 * 2);
    size_t h1   = alloc((size_t)NB * 32 * 400 * 2);
    size_t h2   = alloc((size_t)NB * 64 * 81 * 2);
    size_t h3   = alloc((size_t)NB * 64 * 49 * 2);
    size_t featO= alloc((size_t)NB * 512 * 2);
    size_t hidO = alloc((size_t)NB * 128 * 2);
    size_t valO = alloc((size_t)NB * 4);
    size_t qinO = alloc((size_t)NB * 6 * 4);

    _Float16* p1 = (_Float16*)(ws + wp1);
    _Float16* p2 = (_Float16*)(ws + wp2);
    _Float16* p3 = (_Float16*)(ws + wp3);
    _Float16* pf = (_Float16*)(ws + wpf);
    _Float16* pv = (_Float16*)(ws + wpv);
    _Float16* a1 = (_Float16*)(ws + h1);
    _Float16* a2 = (_Float16*)(ws + h2);
    _Float16* a3 = (_Float16*)(ws + h3);
    _Float16* ft = (_Float16*)(ws + featO);
    _Float16* hd = (_Float16*)(ws + hidO);
    float* valv  = (float*)(ws + valO);
    float* qin   = (float*)(ws + qinO);

    // 1) pack weights into WMMA B-fragment order (ntShift = log2(N/16))
    auto packGrid = [](int K, int ntShift) { return (((K / 32) << (ntShift + 8)) + 255) / 256; };
    pack_w_kernel<<<packGrid(256, 1),  256, 0, stream>>>(conv1_w, p1, 256, 1);
    pack_w_kernel<<<packGrid(512, 2),  256, 0, stream>>>(conv2_w, p2, 512, 2);
    pack_w_kernel<<<packGrid(576, 2),  256, 0, stream>>>(conv3_w, p3, 576, 2);
    pack_w_kernel<<<packGrid(3136, 5), 256, 0, stream>>>(fc_w,    pf, 3136, 5);
    pack_w_kernel<<<packGrid(512, 3),  256, 0, stream>>>(val1_w,  pv, 512, 3);

    // 2) conv1: [1024,4,84,84] -> [1024,32,20,20], M=409600, K=256
    conv_wmma_kernel<true, 2><<<(NB * 400 / 16) / 8, 256, 0, stream>>>(
        x, nullptr, p1, conv1_b, a1, NB, 4, 84, 84, 32, 20, 20, 8, 8, 4);
    // 3) conv2: -> [1024,64,9,9], M=82944, K=512
    conv_wmma_kernel<false, 4><<<(NB * 81 / 16) / 8, 256, 0, stream>>>(
        nullptr, a1, p2, conv2_b, a2, NB, 32, 20, 20, 64, 9, 9, 4, 4, 2);
    // 4) conv3: -> [1024,64,7,7], M=50176, K=576
    conv_wmma_kernel<false, 4><<<(NB * 49 / 16) / 8, 256, 0, stream>>>(
        nullptr, a2, p3, conv3_b, a3, NB, 64, 9, 9, 64, 7, 7, 3, 3, 1);

    // 5) fc: [1024,3136] x [3136,512] -> feat; waves = 64 mtiles * 8 ngroups
    gemm_wmma_kernel<4><<<512 / 8, 256, 0, stream>>>(a3, pf, fc_b, ft, NB, 512, 3136, 3);
    // 6) val1: [1024,512] x [512,128] -> hid; waves = 64 * 2
    gemm_wmma_kernel<4><<<128 / 8, 256, 0, stream>>>(ft, pv, val1_b, hd, NB, 128, 512, 1);

    // 7) heads: value + qin(tanh)
    heads_kernel<<<NB / 256, 256, 0, stream>>>(ft, hd, val2_w, val2_b, adv_w, adv_b, valv, qin);

    // 8) quantum circuit + dueling combine: one wave32 per batch element
    quantum_kernel<<<NB / 8, 256, 0, stream>>>(qin, qparams, qout_w, qout_b, valv, out);
}
